// PointNet2Layer_26688926777504
// MI455X (gfx1250) — matrix operand order repaired and checked by reading, not compiled, not run on hardware
//
#include <hip/hip_runtime.h>

typedef __attribute__((ext_vector_type(16))) _Float16 v16h;
typedef __attribute__((ext_vector_type(8)))  float    v8f;

#define BATCH   4
#define NPTS    16384
#define NPOINT  1024
#define KNB     32
#define RAD2    0.01f
#define CIN     4
#define CH      32

// ---------------------------------------------------------------------------
// Kernel 1: farthest point sampling. One block per batch, 1024 threads.
// Each thread owns 16 points (coords + running min-dist in VGPRs).
// Per round: fused min-update + local argmax, wave32 shuffle butterfly,
// 32-entry LDS cross-wave stage, broadcast next centroid. Writes new_pos.
// ---------------------------------------------------------------------------
#define FPS_T 1024
#define PPT   (NPTS / FPS_T)   // 16

__global__ __launch_bounds__(FPS_T) void fps_kernel(const float* __restrict__ pos,
                                                    float* __restrict__ newpos)
{
  const int b    = blockIdx.x;
  const int t    = threadIdx.x;
  const int lane = t & 31;
  const int wv   = t >> 5;
  const float* p = pos + (size_t)b * NPTS * 3;

  float px[PPT], py[PPT], pz[PPT], mind[PPT];
#pragma unroll
  for (int j = 0; j < PPT; ++j) {
    int n = j * FPS_T + t;
    px[j] = p[n*3+0]; py[j] = p[n*3+1]; pz[j] = p[n*3+2];
    mind[j] = 1e10f;
  }

  __shared__ float scx, scy, scz;
  __shared__ float swv[FPS_T/32];
  __shared__ int   swi[FPS_T/32];
  if (t == 0) { scx = p[0]; scy = p[1]; scz = p[2]; }   // far_0 = 0
  __syncthreads();

  for (int i = 0; i < NPOINT; ++i) {
    const float cx = scx, cy = scy, cz = scz;
    if (t == 0) {
      float* np = newpos + ((size_t)b * NPOINT + i) * 3;
      np[0] = cx; np[1] = cy; np[2] = cz;
    }
    float best = -1.0f; int bi = 0;
#pragma unroll
    for (int j = 0; j < PPT; ++j) {
      float dx = px[j]-cx, dy = py[j]-cy, dz = pz[j]-cz;
      float d  = dx*dx + dy*dy + dz*dz;
      float md = fminf(mind[j], d);
      mind[j]  = md;
      if (md > best) { best = md; bi = j * FPS_T + t; }
    }
    // wave32 argmax butterfly (lower index wins ties, matching jnp.argmax)
#pragma unroll
    for (int off = 16; off > 0; off >>= 1) {
      float ov = __shfl_xor(best, off);
      int   oi = __shfl_xor(bi,   off);
      if (ov > best || (ov == best && oi < bi)) { best = ov; bi = oi; }
    }
    if (lane == 0) { swv[wv] = best; swi[wv] = bi; }
    __syncthreads();
    if (wv == 0) {
      float v  = swv[lane];
      int   ii = swi[lane];
#pragma unroll
      for (int off = 16; off > 0; off >>= 1) {
        float ov = __shfl_xor(v,  off);
        int   oi = __shfl_xor(ii, off);
        if (ov > v || (ov == v && oi < ii)) { v = ov; ii = oi; }
      }
      if (lane == 0) {
        const float* q = p + (size_t)ii * 3;
        scx = q[0]; scy = q[1]; scz = q[2];
      }
    }
    __syncthreads();
  }
}

// ---------------------------------------------------------------------------
// Kernel 2: ball query (k nearest within radius). One wave per centroid.
// Each lane scans a 1/32 stride of N, keeps a sorted <=32 candidate list in
// LDS; then a 32-round wave-wide merge of sorted lists via shuffle
// min-reduction reproduces lax.top_k semantics; pad misses with the nearest.
// ---------------------------------------------------------------------------
#define BQ_WAVES 4

__global__ void ballq_kernel(const float* __restrict__ pos,
                             const float* __restrict__ newpos,
                             int* __restrict__ ballidx)
{
  __shared__ float sd[BQ_WAVES][KNB][32];
  __shared__ int   si[BQ_WAVES][KNB][32];
  const int lane = threadIdx.x & 31;
  const int wv   = threadIdx.x >> 5;
  const int gid  = blockIdx.x * BQ_WAVES + wv;   // [0, B*NPOINT)
  const int b    = gid >> 10;
  const float* p = pos + (size_t)b * NPTS * 3;
  const float cx = newpos[gid*3+0];
  const float cy = newpos[gid*3+1];
  const float cz = newpos[gid*3+2];

  float* myd = &sd[wv][0][lane];   // slot s lives at myd[s*32] (bank-clean)
  int*   myi = &si[wv][0][lane];
  int cnt = 0;

  for (int n = lane; n < NPTS; n += 32) {
    float dx = p[n*3+0]-cx, dy = p[n*3+1]-cy, dz = p[n*3+2]-cz;
    float d2 = dx*dx + dy*dy + dz*dz;
    if (d2 <= RAD2) {
      if (cnt < KNB || d2 < myd[(KNB-1)*32]) {
        int s = (cnt < KNB) ? cnt : (KNB - 1);
        while (s > 0 && myd[(s-1)*32] > d2) {
          myd[s*32] = myd[(s-1)*32];
          myi[s*32] = myi[(s-1)*32];
          --s;
        }
        myd[s*32] = d2; myi[s*32] = n;
        if (cnt < KNB) ++cnt;
      }
    }
  }

  // 32-way merge of per-lane sorted lists
  int ptr = 0, myout = 0, first = 0;
  for (int r = 0; r < KNB; ++r) {
    int   sp = (ptr < KNB) ? ptr : (KNB - 1);
    float v  = (ptr < cnt) ? myd[sp*32] : 3.0e38f;
    int   pi = (ptr < cnt) ? myi[sp*32] : 0x7fffffff;
    int   ln = lane;
#pragma unroll
    for (int off = 16; off > 0; off >>= 1) {
      float ov  = __shfl_xor(v,  off);
      int   opi = __shfl_xor(pi, off);
      int   oln = __shfl_xor(ln, off);
      if (ov < v || (ov == v && opi < pi)) { v = ov; pi = opi; ln = oln; }
    }
    const bool valid = (v <= RAD2);
    if (r == 0)    first = pi;                     // nearest hit (centroid itself)
    if (lane == r) myout = valid ? pi : first;
    if (lane == ln && valid) ++ptr;
  }
  ballidx[(size_t)gid * KNB + lane] = myout;
}

// ---------------------------------------------------------------------------
// Kernel 3: gather + shared MLP + max-pool. One wave per group.
// h staged in per-wave LDS as f16 (K padded 4->32); each layer = 2x2 tiles of
// v_wmma_f32_16x16x32_f16 with fragments packed per the CDNA5 VGPR layouts.
// Final layer fuses bias+ReLU+max-pool; only the pooled [32] vector is stored.
// ---------------------------------------------------------------------------
#define G_WAVES 8
#define HPAD    36

__global__ void group_mlp_kernel(const float* __restrict__ pos,
                                 const float* __restrict__ xf,
                                 const float* __restrict__ W1, const float* __restrict__ b1,
                                 const float* __restrict__ W2, const float* __restrict__ b2,
                                 const float* __restrict__ W3, const float* __restrict__ b3,
                                 const float* __restrict__ newpos,
                                 const int*   __restrict__ ballidx,
                                 float* __restrict__ out)
{
  __shared__ _Float16 hbuf[G_WAVES][32][HPAD];
  const int  lane = threadIdx.x & 31;
  const int  wv   = threadIdx.x >> 5;
  const int  gid  = blockIdx.x * G_WAVES + wv;   // [0, B*NPOINT)
  const int  b    = gid >> 10;
  const int  l16  = lane & 15;
  const bool hiL  = lane >= 16;

  // Weight fragments in WMMA B-layout: lane<16 -> col n=lane, K {0..7,16..23};
  // lane>=16 -> col n=lane-16, K {8..15,24..31}. Layer-1 K rows >=4 are zero.
  const float* Ws[3] = {W1, W2, W3};
  const float* Bs[3] = {b1, b2, b3};
  const int    kd[3] = {CIN, CH, CH};
  v16h  wf[3][2];
  float bv[3][2];
#pragma unroll
  for (int L = 0; L < 3; ++L) {
#pragma unroll
    for (int u = 0; u < 2; ++u) {
      v16h f;
#pragma unroll
      for (int e = 0; e < 16; ++e) {
        int   k = (hiL ? 8 : 0) + e + ((e >= 8) ? 8 : 0);
        float w = (k < kd[L]) ? Ws[L][k*CH + u*16 + l16] : 0.0f;
        f[e] = (_Float16)w;
      }
      wf[L][u] = f;
      bv[L][u] = Bs[L][u*16 + l16];
    }
  }

  // Gather: lane k -> neighbor k of this group; build h0 row (rel-xyz, feat, 0-pad)
  const float npx = newpos[gid*3+0];
  const float npy = newpos[gid*3+1];
  const float npz = newpos[gid*3+2];
  const int   pidx = ballidx[(size_t)gid * KNB + lane];
  const float* pb  = pos + (size_t)b * NPTS * 3;
  const float r0 = pb[pidx*3+0] - npx;
  const float r1 = pb[pidx*3+1] - npy;
  const float r2 = pb[pidx*3+2] - npz;
  const float r3 = xf[(size_t)b * NPTS + pidx];
  _Float16* row = &hbuf[wv][lane][0];
  row[0] = (_Float16)r0; row[1] = (_Float16)r1;
  row[2] = (_Float16)r2; row[3] = (_Float16)r3;
#pragma unroll
  for (int c = CIN; c < CH; ++c) row[c] = (_Float16)0.0f;
  // All LDS traffic is wave-private; DS ops are in-order per wave -> no barrier.

#pragma unroll
  for (int L = 0; L < 3; ++L) {
    // A fragments: lane holds row M = t*16 + (lane&15);
    // lanes<16 take K halves {0..7,16..23}, lanes>=16 take {8..15,24..31}.
    v16h a[2];
#pragma unroll
    for (int t = 0; t < 2; ++t) {
      const int rr = t*16 + l16;
      const int ks = hiL ? 8 : 0;
      v16h av;
#pragma unroll
      for (int e = 0; e < 16; ++e) {
        int col = ks + e + ((e >= 8) ? 8 : 0);
        av[e] = hbuf[wv][rr][col];
      }
      a[t] = av;
    }

    v8f acc[2][2];
    const v8f cz = {0.f,0.f,0.f,0.f,0.f,0.f,0.f,0.f};
#pragma unroll
    for (int t = 0; t < 2; ++t)
#pragma unroll
      for (int u = 0; u < 2; ++u)
        acc[t][u] = __builtin_amdgcn_wmma_f32_16x16x32_f16(
            false, a[t], false, wf[L][u], (short)0, cz, false, false);

    if (L < 2) {
      // bias + ReLU, scatter back into h layout:
      // D: lane<16 -> col n=lane, rows t*16+v; lane>=16 -> col n=lane-16, rows t*16+8+v
#pragma unroll
      for (int t = 0; t < 2; ++t)
#pragma unroll
        for (int u = 0; u < 2; ++u)
#pragma unroll
          for (int v = 0; v < 8; ++v) {
            float val = fmaxf(acc[t][u][v] + bv[L][u], 0.0f);
            hbuf[wv][t*16 + (hiL ? 8 : 0) + v][u*16 + l16] = (_Float16)val;
          }
    } else {
      // fused bias + ReLU + max over K=32 rows (ReLU folded into 0-init of max)
#pragma unroll
      for (int u = 0; u < 2; ++u) {
        float mx = 0.0f;
#pragma unroll
        for (int t = 0; t < 2; ++t)
#pragma unroll
          for (int v = 0; v < 8; ++v)
            mx = fmaxf(mx, acc[t][u][v] + bv[L][u]);
        mx = fmaxf(mx, __shfl_xor(mx, 16));   // combine row halves for this column
        if (!hiL) out[(size_t)gid * CH + u*16 + lane] = mx;
      }
    }
  }
}

// ---------------------------------------------------------------------------
// Launch: fps -> ball query -> group+MLP, all on `stream` (stream-ordered).
// Workspace: new_pos [B,1024,3] f32 (48 KB) + ball idx [B,1024,32] i32 (512 KB).
// ---------------------------------------------------------------------------
extern "C" void kernel_launch(void* const* d_in, const int* in_sizes, int n_in,
                              void* d_out, int out_size, void* d_ws, size_t ws_size,
                              hipStream_t stream)
{
  const float* pos = (const float*)d_in[0];
  const float* xf  = (const float*)d_in[1];
  const float* W1  = (const float*)d_in[2];
  const float* b1  = (const float*)d_in[3];
  const float* W2  = (const float*)d_in[4];
  const float* b2  = (const float*)d_in[5];
  const float* W3  = (const float*)d_in[6];
  const float* b3  = (const float*)d_in[7];
  float* out = (float*)d_out;
  (void)in_sizes; (void)n_in; (void)out_size; (void)ws_size;

  float* newpos  = (float*)d_ws;
  int*   ballidx = (int*)((char*)d_ws + (size_t)BATCH*NPOINT*3*sizeof(float));

  fps_kernel<<<BATCH, FPS_T, 0, stream>>>(pos, newpos);
  ballq_kernel<<<(BATCH*NPOINT)/BQ_WAVES, BQ_WAVES*32, 0, stream>>>(pos, newpos, ballidx);
  group_mlp_kernel<<<(BATCH*NPOINT)/G_WAVES, G_WAVES*32, 0, stream>>>(
      pos, xf, W1, b1, W2, b2, W3, b3, newpos, ballidx, out);
}
